// Transformer_61632780697700
// MI455X (gfx1250) — matrix operand order repaired and checked by reading, not compiled, not run on hardware
//
#include <hip/hip_runtime.h>
#include <hip/hip_bf16.h>
#include <stdint.h>

// ---------------------------------------------------------------------------
// Model dims (compile-time constants from the reference)
// ---------------------------------------------------------------------------
#define BATCH 2
#define SEQ   1024
#define VOCAB 32000
#define DMODEL 512
#define NHEAD 8
#define NLAYER 4
#define FFDIM 2048
#define DHEAD 64
#define WIN 128
#define GLOB 16
#define MROWS (BATCH * SEQ)   // 2048
#define NKB (SEQ / 128)       // 8 key blocks

typedef __attribute__((ext_vector_type(16))) _Float16 v16h;
typedef __attribute__((ext_vector_type(8)))  float    v8f;
typedef __attribute__((ext_vector_type(4)))  unsigned v4u;
typedef __attribute__((ext_vector_type(8)))  unsigned v8u;

// ---------------------------------------------------------------------------
// Band/global mask at 128x128 block granularity:
// element-allowed = |i-j|<=128 || i<16 || j<16 (&& j<=i if causal)
// => block (qb,kb) can contain allowed elements iff qb==0 || kb==0 || |qb-kb|<=1
//    (causal additionally requires kb<=qb).
// ---------------------------------------------------------------------------
__device__ __host__ __forceinline__ bool block_allowed(int qb, int kb, int causal) {
    if (causal && kb > qb) return false;
    return (qb == 0) || (kb == 0) || (qb - kb <= 1 && kb - qb <= 1);
}

// ---------------------------------------------------------------------------
// Helpers: convert 16 contiguous f32 -> 16 f16 in LDS (vectorized b128 loads)
// ---------------------------------------------------------------------------
__device__ __forceinline__ void cvt16_f32_to_h(const float* __restrict__ src,
                                               _Float16* __restrict__ dst) {
#pragma unroll
    for (int i = 0; i < 16; i += 4) {
        float4 f = *(const float4*)(src + i);
        dst[i + 0] = (_Float16)f.x;
        dst[i + 1] = (_Float16)f.y;
        dst[i + 2] = (_Float16)f.z;
        dst[i + 3] = (_Float16)f.w;
    }
}

// A fragment 16x32 f16: lane<16 -> M=lane, K=0..15; lane>=16 -> K=16..31
// LDS layout: row-major [m][k], row stride 32 halves. Two contiguous 16B chunks/lane.
__device__ __forceinline__ v16h load_a_frag(const _Float16* As, int m0, int lane) {
    const _Float16* p = As + (m0 + (lane & 15)) * 32 + ((lane >> 4) << 4);
    v16h a;
#pragma unroll
    for (int i = 0; i < 8; ++i) { a[i] = p[i]; a[i + 8] = p[i + 8]; }
    return a;
}

// B fragment 32x16 f16: lane<16 -> N=lane, K=0..15; lane>=16 -> K=16..31.
// LDS layout: transposed [n][k], row stride 32 halves. One contiguous 32B run/lane.
__device__ __forceinline__ v16h load_b_frag(const _Float16* BsT, int n0, int lane) {
    const _Float16* p = BsT + (n0 + (lane & 15)) * 32 + ((lane >> 4) << 4);
    v16h b;
#pragma unroll
    for (int i = 0; i < 16; ++i) b[i] = p[i];
    return b;
}

__device__ __forceinline__ float gelu_tanh(float x) {
    float x3 = x * x * x;
    return 0.5f * x * (1.0f + tanhf(0.7978845608028654f * (x + 0.044715f * x3)));
}

// ---------------------------------------------------------------------------
// Generic GEMM: C[M,N] = act(A[M,K] @ W[K,N] + bias[N]) (+ residual)
// A,W row-major f32 (lda=K, ldw=N), f16 WMMA with f32 accumulate.
// Block tile 128x128, 8 waves (2x4), each wave 64x32 = 4x2 WMMA accumulators.
// ---------------------------------------------------------------------------
__global__ __launch_bounds__(256)
void gemm_wmma_kernel(const float* __restrict__ A, const float* __restrict__ W,
                      const float* __restrict__ bias, const float* __restrict__ residual,
                      float* __restrict__ C, int M, int N, int K, int act) {
    __shared__ _Float16 As[128 * 32];
    __shared__ _Float16 BsT[128 * 32];

    const int bm = blockIdx.y * 128, bn = blockIdx.x * 128;
    const int tid = threadIdx.x, lane = tid & 31, wid = tid >> 5;
    const int wm = (wid >> 2) * 64, wn = (wid & 3) * 32;

    v8f zero = {};
    v8f acc[4][2];
#pragma unroll
    for (int mi = 0; mi < 4; ++mi)
#pragma unroll
        for (int ni = 0; ni < 2; ++ni) acc[mi][ni] = zero;

    const int am = tid & 127, akc = (tid >> 7) * 16;   // A-tile: 16 halves per thread
    const int bnn = tid & 127, bkc = (tid >> 7) * 16;  // B-tile: 16 halves per thread

    for (int k0 = 0; k0 < K; k0 += 32) {
        cvt16_f32_to_h(A + (size_t)(bm + am) * K + k0 + akc, As + am * 32 + akc);
        {
            const float* src = W + (size_t)(k0 + bkc) * N + bn + bnn;
            _Float16* dst = BsT + bnn * 32 + bkc;
#pragma unroll
            for (int i = 0; i < 16; ++i) dst[i] = (_Float16)src[(size_t)i * N];
        }
        // Prefetch next k-tile into the cache hierarchy (global_prefetch_b8).
        if (k0 + 32 < K) {
            __builtin_prefetch(A + (size_t)(bm + am) * K + k0 + 32 + akc, 0, 1);
            __builtin_prefetch(W + (size_t)(k0 + 32 + bkc) * N + bn + bnn, 0, 1);
        }
        __syncthreads();

        v16h af[4], bf[2];
#pragma unroll
        for (int mi = 0; mi < 4; ++mi) af[mi] = load_a_frag(As, wm + mi * 16, lane);
#pragma unroll
        for (int ni = 0; ni < 2; ++ni) bf[ni] = load_b_frag(BsT, wn + ni * 16, lane);
#pragma unroll
        for (int mi = 0; mi < 4; ++mi)
#pragma unroll
            for (int ni = 0; ni < 2; ++ni)
                acc[mi][ni] = __builtin_amdgcn_wmma_f32_16x16x32_f16(
                    false, af[mi], false, bf[ni], (short)0, acc[mi][ni], false, false);
        __syncthreads();
    }

    const int ml = (lane >> 4) << 3, nl = lane & 15;
#pragma unroll
    for (int mi = 0; mi < 4; ++mi)
#pragma unroll
        for (int ni = 0; ni < 2; ++ni)
#pragma unroll
            for (int r = 0; r < 8; ++r) {
                int row = bm + wm + mi * 16 + ml + r;
                int col = bn + wn + ni * 16 + nl;
                float v = acc[mi][ni][r] + bias[col];
                if (act == 1) v = gelu_tanh(v);
                size_t idx = (size_t)row * N + col;
                if (residual) v += residual[idx];
                C[idx] = v;
            }
}

// ---------------------------------------------------------------------------
// Attention scores: att[bh,q,k] = mask( (Q_bh . K_bh) / 8 ), only for blocks
// that can contain allowed elements. Fully-masked 128x128 tiles are skipped
// entirely (never written, never read downstream).
// grid = (S/128 key-blocks, S/128 query-blocks, B*H)
// ---------------------------------------------------------------------------
__global__ __launch_bounds__(256)
void attn_scores_kernel(const float* __restrict__ Qb, const float* __restrict__ Kb,
                        float* __restrict__ att, int causal) {
    if (!block_allowed(blockIdx.y, blockIdx.x, causal)) return;

    __shared__ _Float16 As[128 * 32];
    __shared__ _Float16 BsT[128 * 32];

    const int bh = blockIdx.z, b = bh >> 3, h = bh & 7;
    const int qm = blockIdx.y * 128, kn = blockIdx.x * 128;
    const float* Qp = Qb + (size_t)b * SEQ * DMODEL + h * DHEAD;
    const float* Kp = Kb + (size_t)b * SEQ * DMODEL + h * DHEAD;

    const int tid = threadIdx.x, lane = tid & 31, wid = tid >> 5;
    const int wm = (wid >> 2) * 64, wn = (wid & 3) * 32;

    v8f zero = {};
    v8f acc[4][2];
#pragma unroll
    for (int mi = 0; mi < 4; ++mi)
#pragma unroll
        for (int ni = 0; ni < 2; ++ni) acc[mi][ni] = zero;

    const int am = tid & 127, akc = (tid >> 7) * 16;

#pragma unroll
    for (int k0 = 0; k0 < DHEAD; k0 += 32) {
        cvt16_f32_to_h(Qp + (size_t)(qm + am) * DMODEL + k0 + akc, As + am * 32 + akc);
        // "B" = K^T : BsT[key][kd] is exactly K row-major -> contiguous convert
        cvt16_f32_to_h(Kp + (size_t)(kn + am) * DMODEL + k0 + akc, BsT + am * 32 + akc);
        __syncthreads();

        v16h af[4], bf[2];
#pragma unroll
        for (int mi = 0; mi < 4; ++mi) af[mi] = load_a_frag(As, wm + mi * 16, lane);
#pragma unroll
        for (int ni = 0; ni < 2; ++ni) bf[ni] = load_b_frag(BsT, wn + ni * 16, lane);
#pragma unroll
        for (int mi = 0; mi < 4; ++mi)
#pragma unroll
            for (int ni = 0; ni < 2; ++ni)
                acc[mi][ni] = __builtin_amdgcn_wmma_f32_16x16x32_f16(
                    false, af[mi], false, bf[ni], (short)0, acc[mi][ni], false, false);
        __syncthreads();
    }

    const int ml = (lane >> 4) << 3, nl = lane & 15;
#pragma unroll
    for (int mi = 0; mi < 4; ++mi)
#pragma unroll
        for (int ni = 0; ni < 2; ++ni)
#pragma unroll
            for (int r = 0; r < 8; ++r) {
                int i = qm + wm + mi * 16 + ml + r;  // query index
                int j = kn + wn + ni * 16 + nl;      // key index
                bool allowed = (abs(i - j) <= WIN) || (i < GLOB) || (j < GLOB);
                if (causal) allowed = allowed && (j <= i);
                float v = allowed ? acc[mi][ni][r] * 0.125f : -1.0e9f;
                att[((size_t)bh * SEQ + i) * SEQ + j] = v;
            }
}

// ---------------------------------------------------------------------------
// Row softmax over the <=4 allowed 128-key tiles. Tiles are staged into LDS
// with the Tensor Data Mover (D# per ISA 8.3/8.4), then reduced from LDS.
// -1e9 entries inside allowed tiles underflow to exactly 0 after exp.
// grid = B*H*S rows, 256 threads.
// ---------------------------------------------------------------------------
__global__ __launch_bounds__(256)
void softmax_kernel(float* __restrict__ att, int causal) {
    __shared__ float rowbuf[512];
    __shared__ float red[256];
    const int row = blockIdx.x;          // bh*SEQ + i
    const int i = row & (SEQ - 1);
    const int qb = i >> 7;
    const size_t base = (size_t)row * SEQ;
    const int t = threadIdx.x;

    int tiles[4]; int nt = 0;
    tiles[nt++] = 0;
    for (int kb = qb - 1; kb <= qb + 1; ++kb)
        if (kb >= 1 && kb <= NKB - 1 && !(causal && kb > qb)) tiles[nt++] = kb;
    const int n = nt * 128;

    // One wave issues the TDM descriptors; whole block consumes after barrier.
    if (t < 32) {
        unsigned ldsbase = (unsigned)(uintptr_t)rowbuf;  // LDS aperture: low 32 bits
        for (int tt = 0; tt < nt; ++tt) {
            unsigned long long ga =
                (unsigned long long)(uintptr_t)(att + base + (size_t)tiles[tt] * 128);
            v4u g0;
            g0.x = 1u;                                  // count=1 (valid descriptor)
            g0.y = ldsbase + (unsigned)tt * 512u;       // lds_addr (bytes)
            g0.z = (unsigned)ga;                        // global_addr[31:0]
            g0.w = (unsigned)(ga >> 32) | 0x80000000u;  // global_addr[56:32] | type=2
            v8u g1;
            g1[0] = 2u << 16;    // data_size = 4 bytes
            g1[1] = 128u << 16;  // tensor_dim0 = 128
            g1[2] = 1u << 16;    // tensor_dim1 = 1
            g1[3] = 128u << 16;  // tile_dim0 = 128
            g1[4] = 1u;          // tile_dim1 = 1
            g1[5] = 128u;        // tensor_dim0_stride
            g1[6] = 0u;
            g1[7] = 0u;
            asm volatile("tensor_load_to_lds %0, %1" :: "s"(g0), "s"(g1) : "memory");
        }
        __builtin_amdgcn_s_wait_tensorcnt(0);
    }
    __syncthreads();

    float m = -3.0e38f;
    for (int e = t; e < n; e += 256) m = fmaxf(m, rowbuf[e]);
    red[t] = m; __syncthreads();
    for (int s = 128; s > 0; s >>= 1) { if (t < s) red[t] = fmaxf(red[t], red[t + s]); __syncthreads(); }
    m = red[0]; __syncthreads();

    float sum = 0.0f;
    for (int e = t; e < n; e += 256) {
        float ev = expf(rowbuf[e] - m);
        rowbuf[e] = ev;         // private slot per thread; re-read by same thread
        sum += ev;
    }
    red[t] = sum; __syncthreads();
    for (int s = 128; s > 0; s >>= 1) { if (t < s) red[t] += red[t + s]; __syncthreads(); }
    const float inv = 1.0f / red[0];

    for (int e = t; e < n; e += 256)
        att[base + (size_t)tiles[e >> 7] * 128 + (e & 127)] = rowbuf[e] * inv;
}

// ---------------------------------------------------------------------------
// Context = A(S x S) @ V(S x DHEAD) per (b,h), iterating only allowed key
// tiles for this query block. Block 128x64, 8 waves (4x2), each wave 32x32.
// grid = (S/128, B*H)
// ---------------------------------------------------------------------------
__global__ __launch_bounds__(256)
void attn_av_kernel(const float* __restrict__ att, const float* __restrict__ Vb,
                    float* __restrict__ ctx, int causal) {
    __shared__ _Float16 As[128 * 32];
    __shared__ _Float16 BsT[64 * 32];

    const int bh = blockIdx.y, b = bh >> 3, h = bh & 7;
    const int qb = blockIdx.x;
    const int qm = qb * 128;
    const float* Ap = att + (size_t)bh * SEQ * SEQ;
    const float* Vp = Vb + (size_t)b * SEQ * DMODEL + h * DHEAD;

    const int tid = threadIdx.x, lane = tid & 31, wid = tid >> 5;
    const int wm = (wid >> 1) * 32, wn = (wid & 1) * 32;

    int tiles[4]; int nt = 0;
    tiles[nt++] = 0;
    for (int kb = qb - 1; kb <= qb + 1; ++kb)
        if (kb >= 1 && kb <= NKB - 1 && !(causal && kb > qb)) tiles[nt++] = kb;

    v8f zero = {};
    v8f acc[2][2];
#pragma unroll
    for (int mi = 0; mi < 2; ++mi)
#pragma unroll
        for (int ni = 0; ni < 2; ++ni) acc[mi][ni] = zero;

    const int am = tid & 127, akc = (tid >> 7) * 16;
    const int vn = tid & 63, vkc = (tid >> 6) * 8;

    for (int tt = 0; tt < nt; ++tt) {
        const int kb = tiles[tt];
#pragma unroll
        for (int ks = 0; ks < 4; ++ks) {
            const int k0 = kb * 128 + ks * 32;
            cvt16_f32_to_h(Ap + (size_t)(qm + am) * SEQ + k0 + akc, As + am * 32 + akc);
            {
                _Float16* dst = BsT + vn * 32 + vkc;
#pragma unroll
                for (int i = 0; i < 8; ++i)
                    dst[i] = (_Float16)Vp[(size_t)(k0 + vkc + i) * DMODEL + vn];
            }
            __syncthreads();

            v16h af[2], bf[2];
#pragma unroll
            for (int mi = 0; mi < 2; ++mi) af[mi] = load_a_frag(As, wm + mi * 16, lane);
#pragma unroll
            for (int ni = 0; ni < 2; ++ni) bf[ni] = load_b_frag(BsT, wn + ni * 16, lane);
#pragma unroll
            for (int mi = 0; mi < 2; ++mi)
#pragma unroll
                for (int ni = 0; ni < 2; ++ni)
                    acc[mi][ni] = __builtin_amdgcn_wmma_f32_16x16x32_f16(
                        false, af[mi], false, bf[ni], (short)0, acc[mi][ni], false, false);
            __syncthreads();
        }
    }

    const int ml = (lane >> 4) << 3, nl = lane & 15;
#pragma unroll
    for (int mi = 0; mi < 2; ++mi)
#pragma unroll
        for (int ni = 0; ni < 2; ++ni)
#pragma unroll
            for (int r = 0; r < 8; ++r) {
                int row = qm + wm + mi * 16 + ml + r;
                int col = wn + ni * 16 + nl;
                ctx[((size_t)b * SEQ + row) * DMODEL + h * DHEAD + col] = acc[mi][ni][r];
            }
}

// ---------------------------------------------------------------------------
// Elementwise kernels
// ---------------------------------------------------------------------------
__global__ __launch_bounds__(256)
void embed_kernel(const int* __restrict__ tok, const float* __restrict__ table,
                  float* __restrict__ out) {
    const size_t row = blockIdx.x;
    const int tk = tok[row];
    const float* src = table + (size_t)tk * DMODEL;
    float* dst = out + row * DMODEL;
    for (int i = threadIdx.x; i < DMODEL; i += blockDim.x)
        dst[i] = src[i] * 22.627416997969522f;  // sqrt(512)
}

__global__ __launch_bounds__(256)
void layernorm_kernel(const float* __restrict__ x, const float* __restrict__ gamma,
                      const float* __restrict__ beta, float* __restrict__ out) {
    __shared__ float red[256];
    const size_t base = (size_t)blockIdx.x * DMODEL;
    const int t = threadIdx.x;
    float a = x[base + t], b = x[base + t + 256];

    red[t] = a + b; __syncthreads();
    for (int s = 128; s > 0; s >>= 1) { if (t < s) red[t] += red[t + s]; __syncthreads(); }
    float mean = red[0] * (1.0f / DMODEL); __syncthreads();

    float da = a - mean, db = b - mean;
    red[t] = da * da + db * db; __syncthreads();
    for (int s = 128; s > 0; s >>= 1) { if (t < s) red[t] += red[t + s]; __syncthreads(); }
    float rstd = rsqrtf(red[0] * (1.0f / DMODEL) + 1e-5f);

    out[base + t]       = gamma[t] * da * rstd + beta[t];
    out[base + t + 256] = gamma[t + 256] * db * rstd + beta[t + 256];
}

// RoPE in-place on (B*S, D) buffer; one thread per (row, head, d<32)
__global__ __launch_bounds__(256)
void rope_kernel(float* __restrict__ x) {
    const int idx = blockIdx.x * blockDim.x + threadIdx.x;  // B*S*H*32 total
    const int d = idx & 31;
    const int h = (idx >> 5) & 7;
    const size_t row = (size_t)(idx >> 8);
    const int pos = (int)(row & (SEQ - 1));
    float inv = __powf(10000.0f, -(float)d * (1.0f / 32.0f));
    float ang = (float)pos * inv;
    float c = __cosf(ang), sn = __sinf(ang);
    size_t p = row * DMODEL + h * DHEAD + d;
    float x1 = x[p], x2 = x[p + 32];
    x[p]      = x1 * c - x2 * sn;
    x[p + 32] = x1 * sn + x2 * c;
}

// ---------------------------------------------------------------------------
// Host orchestration
// ---------------------------------------------------------------------------
extern "C" void kernel_launch(void* const* d_in, const int* in_sizes, int n_in,
                              void* d_out, int out_size, void* d_ws, size_t ws_size,
                              hipStream_t stream) {
    (void)in_sizes; (void)n_in; (void)out_size; (void)ws_size;

    const int*   src         = (const int*)  d_in[0];
    const int*   tgt         = (const int*)  d_in[1];
    const float* enc_embed   = (const float*)d_in[2];
    const float* dec_embed   = (const float*)d_in[3];
    const float* enc_attn_w  = (const float*)d_in[4];
    const float* enc_attn_b  = (const float*)d_in[5];
    const float* enc_ln      = (const float*)d_in[6];
    const float* enc_ffn_w1  = (const float*)d_in[7];
    const float* enc_ffn_b1  = (const float*)d_in[8];
    const float* enc_ffn_w2  = (const float*)d_in[9];
    const float* enc_ffn_b2  = (const float*)d_in[10];
    const float* enc_fln     = (const float*)d_in[11];
    const float* dec_attn_w  = (const float*)d_in[12];
    const float* dec_attn_b  = (const float*)d_in[13];
    const float* dec_ln      = (const float*)d_in[14];
    const float* dec_ffn_w1  = (const float*)d_in[15];
    const float* dec_ffn_b1  = (const float*)d_in[16];
    const float* dec_ffn_w2  = (const float*)d_in[17];
    const float* dec_ffn_b2  = (const float*)d_in[18];
    const float* dec_fln     = (const float*)d_in[19];
    const float* out_w       = (const float*)d_in[20];
    const float* out_b       = (const float*)d_in[21];

    const int D = DMODEL, FF = FFDIM, M = MROWS, V = VOCAB;

    // Workspace layout (floats)
    float* ws = (float*)d_ws;
    const size_t SXD = (size_t)M * D;             // 2048*512
    float* X   = ws;
    float* Y   = X + SXD;
    float* MEMB = Y + SXD;
    float* NB  = MEMB + SXD;
    float* Qb  = NB + SXD;
    float* Kb  = Qb + SXD;
    float* Vb  = Kb + SXD;
    float* CTX = Vb + SXD;
    float* HID = CTX + SXD;                        // 2048*2048
    float* ATT = HID + (size_t)M * FF;             // 2*8*1024*1024

    auto gemm = [&](const float* A, const float* W, const float* bias, const float* res,
                    float* C, int mm, int nn, int kk, int act) {
        dim3 grid(nn / 128, mm / 128);
        gemm_wmma_kernel<<<grid, 256, 0, stream>>>(A, W, bias, res, C, mm, nn, kk, act);
    };
    auto ln = [&](const float* xin, const float* g, const float* b, float* o) {
        layernorm_kernel<<<M, 256, 0, stream>>>(xin, g, b, o);
    };
    auto attention = [&](const float* qin, const float* kvin, const float* w,
                         const float* bvec, float* xio, int causal) {
        gemm(qin,  w + 0 * (size_t)D * D, bvec + 0 * D, nullptr, Qb, M, D, D, 0);
        gemm(kvin, w + 1 * (size_t)D * D, bvec + 1 * D, nullptr, Kb, M, D, D, 0);
        gemm(kvin, w + 2 * (size_t)D * D, bvec + 2 * D, nullptr, Vb, M, D, D, 0);
        rope_kernel<<<(M * NHEAD * 32) / 256, 256, 0, stream>>>(Qb);
        rope_kernel<<<(M * NHEAD * 32) / 256, 256, 0, stream>>>(Kb);
        attn_scores_kernel<<<dim3(NKB, NKB, BATCH * NHEAD), 256, 0, stream>>>(
            Qb, Kb, ATT, causal);
        softmax_kernel<<<BATCH * NHEAD * SEQ, 256, 0, stream>>>(ATT, causal);
        attn_av_kernel<<<dim3(NKB, BATCH * NHEAD), 256, 0, stream>>>(ATT, Vb, CTX, causal);
        gemm(CTX, w + 3 * (size_t)D * D, bvec + 3 * D, xio, xio, M, D, D, 0);
    };
    auto ffn = [&](const float* n, const float* w1, const float* b1,
                   const float* w2, const float* b2, float* xio) {
        gemm(n, w1, b1, nullptr, HID, M, FF, D, 1);   // fused GELU
        gemm(HID, w2, b2, xio, xio, M, D, FF, 0);     // fused residual
    };

    // -------------------- Encoder --------------------
    embed_kernel<<<M, 256, 0, stream>>>(src, enc_embed, X);
    for (int l = 0; l < NLAYER; ++l) {
        const float* w   = enc_attn_w + (size_t)l * 4 * D * D;
        const float* bb  = enc_attn_b + (size_t)l * 4 * D;
        const float* lnp = enc_ln + (size_t)l * 2 * 2 * D;  // [which][g/b][D]
        ln(X, lnp, lnp + D, NB);
        attention(NB, NB, w, bb, X, /*causal=*/0);
        ln(X, lnp + 2 * D, lnp + 3 * D, NB);
        ffn(NB, enc_ffn_w1 + (size_t)l * D * FF, enc_ffn_b1 + (size_t)l * FF,
            enc_ffn_w2 + (size_t)l * FF * D, enc_ffn_b2 + (size_t)l * D, X);
    }
    ln(X, enc_fln, enc_fln + D, MEMB);

    // -------------------- Decoder --------------------
    embed_kernel<<<M, 256, 0, stream>>>(tgt, dec_embed, Y);
    for (int l = 0; l < NLAYER; ++l) {
        const float* w   = dec_attn_w + (size_t)l * 8 * D * D;
        const float* bb  = dec_attn_b + (size_t)l * 8 * D;
        const float* lnp = dec_ln + (size_t)l * 3 * 2 * D;
        ln(Y, lnp, lnp + D, NB);
        attention(NB, NB, w, bb, Y, /*causal=*/1);              // self
        ln(Y, lnp + 2 * D, lnp + 3 * D, NB);
        attention(NB, MEMB, w + 4 * (size_t)D * D, bb + 4 * D, Y, /*causal=*/0);  // cross
        ln(Y, lnp + 4 * D, lnp + 5 * D, NB);
        ffn(NB, dec_ffn_w1 + (size_t)l * D * FF, dec_ffn_b1 + (size_t)l * FF,
            dec_ffn_w2 + (size_t)l * FF * D, dec_ffn_b2 + (size_t)l * D, Y);
    }
    ln(Y, dec_fln, dec_fln + D, NB);

    // -------------------- Output projection --------------------
    gemm(NB, out_w, out_b, nullptr, (float*)d_out, M, V, D, 0);  // 2048 x 32000
}
